// AttGCNN_23948737642597
// MI455X (gfx1250) — matrix operand order repaired and checked by reading, not compiled, not run on hardware
//
#include <hip/hip_runtime.h>
#include <hip/hip_bf16.h>

// ---------------------------------------------------------------------------
// GATv2 x2 forward for MI455X (gfx1250, wave32).
//
// Pipeline (all on `stream`, workspace laid out below, ~104 MB):
//   1. zero-init atomics region
//   2. pack Wl1/Wr1 (f32) -> bf16 WMMA B-fragment layout       (once, 64 KB)
//   3. convert x -> bf16                                       (12.8 MB)
//   4. WMMA GEMM: xl1 = x@Wl1+bl1, xr1 = x@Wr1+br1  (v_wmma_f32_16x16x32_bf16)
//   5. build CSR over dst (count / block-scan / scatter)
//   6. edge pass: alpha[e][h] (leaky 0.2 + att dot), atomic-max per dst
//   7. edge pass: ex = exp(alpha - amax[dst]), atomic-add denom
//   8. node pass (wave32 per node): out = sum a*xl1[src] + bias1, leaky 0.01
//   9. layer-2 transforms (K=128 -> 4, VALU)
//  10/11. layer-2 edge passes (alpha2 / ex2)
//  12. node pass: aggregate 4ch + bias2, softmax -> d_out
// ---------------------------------------------------------------------------

#define N_NODES 50000
#define N_EDGES 600000
#define E_TOT   650000          // edges + self loops
#define TILES_M 3125            // 50000 / 16

typedef __attribute__((ext_vector_type(16))) __bf16 v16bf;
typedef __attribute__((ext_vector_type(8)))  __bf16 v8bf;
typedef __attribute__((ext_vector_type(8)))  float  v8f;

static __device__ __forceinline__ int srcOf(const long long* ei, int e) {
  return (e < N_EDGES) ? (int)ei[e] : (e - N_EDGES);
}
static __device__ __forceinline__ int dstOf(const long long* ei, int e) {
  return (e < N_EDGES) ? (int)ei[N_EDGES + e] : (e - N_EDGES);
}
// order-preserving float<->uint map so atomicMax(u32) implements float max
static __device__ __forceinline__ unsigned encf(float f) {
  unsigned u = __float_as_uint(f);
  return (u & 0x80000000u) ? ~u : (u | 0x80000000u);
}
static __device__ __forceinline__ float decf(unsigned v) {
  unsigned u = (v & 0x80000000u) ? (v ^ 0x80000000u) : ~v;
  return __uint_as_float(u);
}
static __device__ __forceinline__ float lrelu(float v, float s) {
  return v > 0.0f ? v : s * v;
}

// --------------------------- 1. zero init ----------------------------------
__global__ void k_zero(unsigned* __restrict__ p, int n) {
  int i = blockIdx.x * blockDim.x + threadIdx.x;
  if (i < n) p[i] = 0u;
}

// ------------------- 2. pack weights into B-fragment layout ----------------
// layout: [mat(2)][ct(8)][t(4)][lane(32)][i(16)] bf16
// B element (k, n), kk = k mod 32: lane = (n&15) + 16*((kk>>3)&1), i = (kk&7) + 8*(kk>>4)
__global__ void k_pack_w(const float* __restrict__ Wl,
                         const float* __restrict__ Wr,
                         unsigned short* __restrict__ wpack_us) {
  int idx = blockIdx.x * blockDim.x + threadIdx.x;
  if (idx >= 2 * 16384) return;
  int i    = idx & 15;
  int lane = (idx >> 4) & 31;
  int t    = (idx >> 9) & 3;
  int ct   = (idx >> 11) & 7;
  int mat  = idx >> 14;
  int k = t * 32 + 8 * (lane >> 4) + (i & 7) + 16 * (i >> 3);
  int n = ct * 16 + (lane & 15);
  const float* W = mat ? Wr : Wl;
  ((__bf16*)wpack_us)[idx] = (__bf16)W[k * 128 + n];
}

// --------------------------- 3. x -> bf16 ----------------------------------
__global__ void k_x_to_bf16(const float* __restrict__ x,
                            unsigned short* __restrict__ xbf_us, int n) {
  int i = blockIdx.x * blockDim.x + threadIdx.x;
  if (i < n) ((__bf16*)xbf_us)[i] = (__bf16)x[i];
}

// --------------------- 4. layer-1 GEMMs via WMMA ---------------------------
// one wave per 16-row tile; computes both xl1 and xr1 (A fragments reused).
// Per column tile: all 4 B fragments are preloaded (one 8x b128 clause) before
// the 4-deep WMMA chain, and bias values are hoisted to kernel entry, so the
// store sequence never stalls on a fresh load.
__global__ void k_gemm1(const unsigned short* __restrict__ xbf_us,
                        const unsigned short* __restrict__ wpack_us,
                        const float* __restrict__ bl, const float* __restrict__ br,
                        float* __restrict__ xl1, float* __restrict__ xr1) {
  const int lane = threadIdx.x & 31;
  const int wave = (blockIdx.x * blockDim.x + threadIdx.x) >> 5;
  if (wave >= TILES_M) return;           // wave-uniform: EXEC stays all-ones
  const int m0 = wave << 4;

  const __bf16* xb = (const __bf16*)xbf_us;
  const __bf16* wp = (const __bf16*)wpack_us;

  // hoisted bias values for this lane's output column in each column tile
  float blv[8], brv[8];
#pragma unroll
  for (int ct = 0; ct < 8; ++ct) {
    blv[ct] = bl[ct * 16 + (lane & 15)];
    brv[ct] = br[ct * 16 + (lane & 15)];
  }

  // A fragments: lane (m = lane&15) holds K runs [koff, koff+8) and [koff+16, koff+24)
  const int koff = (lane >> 4) << 3;
  const __bf16* xrow = xb + (m0 + (lane & 15)) * 128 + koff;
  v16bf a[4];
#pragma unroll
  for (int t = 0; t < 4; ++t) {
    v8bf lo = *(const v8bf*)(xrow + 32 * t);
    v8bf hi = *(const v8bf*)(xrow + 32 * t + 16);
    a[t] = __builtin_shufflevector(lo, hi, 0, 1, 2, 3, 4, 5, 6, 7,
                                   8, 9, 10, 11, 12, 13, 14, 15);
  }

  const int mbase = m0 + ((lane >> 4) << 3);   // D: vgpr r -> row r + 8*half
#pragma unroll
  for (int mat = 0; mat < 2; ++mat) {
    const __bf16* wm = wp + mat * 16384 + lane * 16;
    const float* biasv = mat ? brv : blv;
    float* out = mat ? xr1 : xl1;
#pragma unroll
    for (int ct = 0; ct < 8; ++ct) {
      // preload all 4 B fragments (8x global_load_b128, one clause)
      v16bf b[4];
#pragma unroll
      for (int t = 0; t < 4; ++t)
        b[t] = *(const v16bf*)(wm + ct * 2048 + t * 512);
      v8f c = {};
#pragma unroll
      for (int t = 0; t < 4; ++t)
        c = __builtin_amdgcn_wmma_f32_16x16x32_bf16(false, a[t], false, b[t],
                                                    (short)0, c, false, false);
      const int col = ct * 16 + (lane & 15);
      const float bv = biasv[ct];
#pragma unroll
      for (int r = 0; r < 8; ++r)
        out[(mbase + r) * 128 + col] = c[r] + bv;
    }
  }
}

// --------------------------- 5. CSR build ----------------------------------
__global__ void k_count_deg(const long long* __restrict__ ei,
                            unsigned* __restrict__ deg) {
  int e = blockIdx.x * blockDim.x + threadIdx.x;
  if (e < E_TOT) atomicAdd(&deg[dstOf(ei, e)], 1u);
}

// single 1024-thread block: exclusive scan of deg -> rowstart[0..N]
__global__ void k_scan(const unsigned* __restrict__ deg,
                       unsigned* __restrict__ rowstart) {
  __shared__ unsigned sbuf[1024];
  __shared__ unsigned carry;
  if (threadIdx.x == 0) carry = 0u;
  __syncthreads();
  for (int base = 0; base < N_NODES; base += 1024) {
    int i = base + (int)threadIdx.x;
    unsigned v = (i < N_NODES) ? deg[i] : 0u;
    sbuf[threadIdx.x] = v;
    __syncthreads();
    for (int s = 1; s < 1024; s <<= 1) {
      unsigned add = (threadIdx.x >= (unsigned)s) ? sbuf[threadIdx.x - s] : 0u;
      __syncthreads();
      sbuf[threadIdx.x] += add;
      __syncthreads();
    }
    unsigned incl = sbuf[threadIdx.x];
    unsigned tot  = sbuf[1023];
    unsigned cpre = carry;
    __syncthreads();
    if (i < N_NODES) rowstart[i] = cpre + incl - v;
    if (threadIdx.x == 0) carry = cpre + tot;
    __syncthreads();
  }
  if (threadIdx.x == 0) rowstart[N_NODES] = carry;
}

__global__ void k_scatter(const long long* __restrict__ ei,
                          const unsigned* __restrict__ rowstart,
                          unsigned* __restrict__ cursor,
                          unsigned* __restrict__ csr) {
  int e = blockIdx.x * blockDim.x + threadIdx.x;
  if (e >= E_TOT) return;
  int d = dstOf(ei, e);
  unsigned pos = atomicAdd(&cursor[d], 1u);
  csr[rowstart[d] + pos] = (unsigned)e;
}

// ---------------- 6. layer-1 edge alpha + segment max ----------------------
__global__ void k_edge1_alpha(const long long* __restrict__ ei,
                              const float* __restrict__ xl1,
                              const float* __restrict__ xr1,
                              const float* __restrict__ att1,
                              float* __restrict__ alpha1,
                              unsigned* __restrict__ amax1) {
  int e = blockIdx.x * blockDim.x + threadIdx.x;
  if (e >= E_TOT) return;
  int s = srcOf(ei, e), d = dstOf(ei, e);
  const float4* pa = (const float4*)(xl1 + (size_t)s * 128);
  const float4* pb = (const float4*)(xr1 + (size_t)d * 128);
  const float4* pw = (const float4*)att1;   // att1 flat [2*64] == channel index
  float acc0 = 0.0f, acc1 = 0.0f;
#pragma unroll 4
  for (int q = 0; q < 32; ++q) {
    float4 u = pa[q], v = pb[q], w = pw[q];
    float e0 = lrelu(u.x + v.x, 0.2f);
    float e1 = lrelu(u.y + v.y, 0.2f);
    float e2 = lrelu(u.z + v.z, 0.2f);
    float e3 = lrelu(u.w + v.w, 0.2f);
    float dt = e0 * w.x + e1 * w.y + e2 * w.z + e3 * w.w;
    if (q < 16) acc0 += dt; else acc1 += dt;
  }
  alpha1[e * 2 + 0] = acc0;
  alpha1[e * 2 + 1] = acc1;
  atomicMax(&amax1[d * 2 + 0], encf(acc0));
  atomicMax(&amax1[d * 2 + 1], encf(acc1));
}

// ---------------- 7. layer-1 exp + segment denom ---------------------------
__global__ void k_edge1_exp(const long long* __restrict__ ei,
                            float* __restrict__ alpha_ex,   // in: alpha, out: ex
                            const unsigned* __restrict__ amax1,
                            float* __restrict__ denom1) {
  int e = blockIdx.x * blockDim.x + threadIdx.x;
  if (e >= E_TOT) return;
  int d = dstOf(ei, e);
#pragma unroll
  for (int h = 0; h < 2; ++h) {
    float ex = expf(alpha_ex[e * 2 + h] - decf(amax1[d * 2 + h]));
    alpha_ex[e * 2 + h] = ex;
    atomicAdd(&denom1[d * 2 + h], ex);
  }
}

// -------- 8. layer-1 aggregation: wave32 per node, 4 channels/lane ---------
__global__ void k_node1_aggr(const long long* __restrict__ ei,
                             const unsigned* __restrict__ rowstart,
                             const unsigned* __restrict__ csr,
                             const float* __restrict__ ex1,
                             const float* __restrict__ denom1,
                             const float* __restrict__ xl1,
                             const float* __restrict__ bias1,
                             float* __restrict__ hbuf) {
  const int lane = threadIdx.x & 31;
  const int n = (blockIdx.x * blockDim.x + threadIdx.x) >> 5;
  if (n >= N_NODES) return;
  const int c0 = lane << 2;
  const int head = c0 >> 6;
  const float inv = 1.0f / (denom1[n * 2 + head] + 1e-16f);
  float4 acc = make_float4(0.f, 0.f, 0.f, 0.f);
  const unsigned b0 = rowstart[n], b1 = rowstart[n + 1];
  for (unsigned idx = b0; idx < b1; ++idx) {
    int e = (int)csr[idx];
    int s = srcOf(ei, e);
    float a = ex1[e * 2 + head] * inv;
    float4 v = ((const float4*)(xl1 + (size_t)s * 128))[lane];
    acc.x += a * v.x; acc.y += a * v.y; acc.z += a * v.z; acc.w += a * v.w;
  }
  float4 bz = ((const float4*)bias1)[lane];
  float4 o;
  o.x = lrelu(acc.x + bz.x, 0.01f);
  o.y = lrelu(acc.y + bz.y, 0.01f);
  o.z = lrelu(acc.z + bz.z, 0.01f);
  o.w = lrelu(acc.w + bz.w, 0.01f);
  ((float4*)(hbuf + (size_t)n * 128))[lane] = o;
}

// ------------------ 9. layer-2 transforms (128 -> 4, VALU) -----------------
__global__ void k_gemm2(const float* __restrict__ hbuf,
                        const float* __restrict__ Wl2, const float* __restrict__ bl2,
                        const float* __restrict__ Wr2, const float* __restrict__ br2,
                        float* __restrict__ xl2, float* __restrict__ xr2) {
  int n = blockIdx.x * blockDim.x + threadIdx.x;
  if (n >= N_NODES) return;
  const float* hr = hbuf + (size_t)n * 128;
  float4 al = make_float4(0.f, 0.f, 0.f, 0.f);
  float4 ar = make_float4(0.f, 0.f, 0.f, 0.f);
#pragma unroll 8
  for (int k = 0; k < 128; ++k) {
    float hv = hr[k];
    float4 wl = ((const float4*)Wl2)[k];
    float4 wr = ((const float4*)Wr2)[k];
    al.x += hv * wl.x; al.y += hv * wl.y; al.z += hv * wl.z; al.w += hv * wl.w;
    ar.x += hv * wr.x; ar.y += hv * wr.y; ar.z += hv * wr.z; ar.w += hv * wr.w;
  }
  float4 b1 = *(const float4*)bl2;
  float4 b2 = *(const float4*)br2;
  ((float4*)xl2)[n] = make_float4(al.x + b1.x, al.y + b1.y, al.z + b1.z, al.w + b1.w);
  ((float4*)xr2)[n] = make_float4(ar.x + b2.x, ar.y + b2.y, ar.z + b2.z, ar.w + b2.w);
}

// ------------------------ 10/11. layer-2 edge passes -----------------------
__global__ void k_edge2_alpha(const long long* __restrict__ ei,
                              const float* __restrict__ xl2,
                              const float* __restrict__ xr2,
                              const float* __restrict__ att2,
                              float* __restrict__ alpha2,
                              unsigned* __restrict__ amax2) {
  int e = blockIdx.x * blockDim.x + threadIdx.x;
  if (e >= E_TOT) return;
  int s = srcOf(ei, e), d = dstOf(ei, e);
  float4 u = ((const float4*)xl2)[s];
  float4 v = ((const float4*)xr2)[d];
  float4 w = *(const float4*)att2;
  float a = lrelu(u.x + v.x, 0.2f) * w.x + lrelu(u.y + v.y, 0.2f) * w.y +
            lrelu(u.z + v.z, 0.2f) * w.z + lrelu(u.w + v.w, 0.2f) * w.w;
  alpha2[e] = a;
  atomicMax(&amax2[d], encf(a));
}

__global__ void k_edge2_exp(const long long* __restrict__ ei,
                            float* __restrict__ alpha_ex2,
                            const unsigned* __restrict__ amax2,
                            float* __restrict__ denom2) {
  int e = blockIdx.x * blockDim.x + threadIdx.x;
  if (e >= E_TOT) return;
  int d = dstOf(ei, e);
  float ex = expf(alpha_ex2[e] - decf(amax2[d]));
  alpha_ex2[e] = ex;
  atomicAdd(&denom2[d], ex);
}

// -------------- 12. layer-2 aggregation + softmax -> d_out -----------------
__global__ void k_node2_out(const long long* __restrict__ ei,
                            const unsigned* __restrict__ rowstart,
                            const unsigned* __restrict__ csr,
                            const float* __restrict__ ex2,
                            const float* __restrict__ denom2,
                            const float* __restrict__ xl2,
                            const float* __restrict__ bias2,
                            float* __restrict__ out) {
  int n = blockIdx.x * blockDim.x + threadIdx.x;
  if (n >= N_NODES) return;
  const float inv = 1.0f / (denom2[n] + 1e-16f);
  float4 acc = make_float4(0.f, 0.f, 0.f, 0.f);
  const unsigned b0 = rowstart[n], b1 = rowstart[n + 1];
  for (unsigned idx = b0; idx < b1; ++idx) {
    int e = (int)csr[idx];
    int s = srcOf(ei, e);
    float a = ex2[e] * inv;
    float4 v = ((const float4*)xl2)[s];
    acc.x += a * v.x; acc.y += a * v.y; acc.z += a * v.z; acc.w += a * v.w;
  }
  float4 bz = *(const float4*)bias2;
  float t0 = acc.x + bz.x, t1 = acc.y + bz.y, t2 = acc.z + bz.z, t3 = acc.w + bz.w;
  float m = fmaxf(fmaxf(t0, t1), fmaxf(t2, t3));
  float e0 = expf(t0 - m), e1 = expf(t1 - m), e2 = expf(t2 - m), e3 = expf(t3 - m);
  float is = 1.0f / (e0 + e1 + e2 + e3);
  out[n * 4 + 0] = e0 * is;
  out[n * 4 + 1] = e1 * is;
  out[n * 4 + 2] = e2 * is;
  out[n * 4 + 3] = e3 * is;
}

// ---------------------------------------------------------------------------
extern "C" void kernel_launch(void* const* d_in, const int* in_sizes, int n_in,
                              void* d_out, int out_size, void* d_ws, size_t ws_size,
                              hipStream_t stream) {
  (void)in_sizes; (void)n_in; (void)out_size; (void)ws_size;
  const float*     x     = (const float*)d_in[0];
  const long long* ei    = (const long long*)d_in[1];   // edge_index int64 [2,E]
  /* d_in[2] = weights: unused by the reference */
  const float* Wl1  = (const float*)d_in[3];
  const float* bl1  = (const float*)d_in[4];
  const float* Wr1  = (const float*)d_in[5];
  const float* br1  = (const float*)d_in[6];
  const float* att1 = (const float*)d_in[7];
  const float* bias1= (const float*)d_in[8];
  const float* Wl2  = (const float*)d_in[9];
  const float* bl2  = (const float*)d_in[10];
  const float* Wr2  = (const float*)d_in[11];
  const float* br2  = (const float*)d_in[12];
  const float* att2 = (const float*)d_in[13];
  const float* bias2= (const float*)d_in[14];
  float* out = (float*)d_out;

  // ---- workspace layout (~104 MB total) ----
  char* ws = (char*)d_ws;
  size_t off = 0;
  auto take = [&](size_t bytes) -> char* {
    char* p = ws + off;
    off = (off + bytes + 255) & ~(size_t)255;
    return p;
  };
  // contiguous zero-init region (atomics): 400000 u32 words
  unsigned* zbase  = (unsigned*)take(400000u * 4);
  unsigned* deg    = zbase;                       // [N]
  unsigned* cursor = zbase + 50000;               // [N]
  unsigned* amax1  = zbase + 100000;              // [N*2] encoded float max
  float*    denom1 = (float*)(zbase + 200000);    // [N*2]
  unsigned* amax2  = zbase + 300000;              // [N]
  float*    denom2 = (float*)(zbase + 350000);    // [N]
  unsigned* rowstart = (unsigned*)take((N_NODES + 1) * 4u);
  unsigned* csr      = (unsigned*)take((size_t)E_TOT * 4u);
  unsigned short* xbf   = (unsigned short*)take((size_t)N_NODES * 128 * 2);
  unsigned short* wpack = (unsigned short*)take(32768u * 2);
  float* xl1  = (float*)take((size_t)N_NODES * 128 * 4);
  float* xr1  = (float*)take((size_t)N_NODES * 128 * 4);
  float* hbuf = (float*)take((size_t)N_NODES * 128 * 4);
  float* ex1  = (float*)take((size_t)E_TOT * 2 * 4);   // alpha then ex, in place
  float* xl2  = (float*)take((size_t)N_NODES * 4 * 4);
  float* xr2  = (float*)take((size_t)N_NODES * 4 * 4);
  float* ex2  = (float*)take((size_t)E_TOT * 4);       // alpha2 then ex2

  const int TB = 256;
  const int gbE = (E_TOT + TB - 1) / TB;
  const int gbN = (N_NODES + TB - 1) / TB;

  k_zero<<<(400000 + TB - 1) / TB, TB, 0, stream>>>(zbase, 400000);
  k_pack_w<<<(2 * 16384 + TB - 1) / TB, TB, 0, stream>>>(Wl1, Wr1, wpack);
  k_x_to_bf16<<<(N_NODES * 128 + TB - 1) / TB, TB, 0, stream>>>(x, xbf, N_NODES * 128);
  // 3125 waves, 8 waves (256 threads) per block
  k_gemm1<<<(TILES_M + 7) / 8, TB, 0, stream>>>(xbf, wpack, bl1, br1, xl1, xr1);
  k_count_deg<<<gbE, TB, 0, stream>>>(ei, deg);
  k_scan<<<1, 1024, 0, stream>>>(deg, rowstart);
  k_scatter<<<gbE, TB, 0, stream>>>(ei, rowstart, cursor, csr);
  k_edge1_alpha<<<gbE, TB, 0, stream>>>(ei, xl1, xr1, att1, ex1, amax1);
  k_edge1_exp<<<gbE, TB, 0, stream>>>(ei, ex1, amax1, denom1);
  k_node1_aggr<<<(N_NODES * 32 + TB - 1) / TB, TB, 0, stream>>>(
      ei, rowstart, csr, ex1, denom1, xl1, bias1, hbuf);
  k_gemm2<<<gbN, TB, 0, stream>>>(hbuf, Wl2, bl2, Wr2, br2, xl2, xr2);
  k_edge2_alpha<<<gbE, TB, 0, stream>>>(ei, xl2, xr2, att2, ex2, amax2);
  k_edge2_exp<<<gbE, TB, 0, stream>>>(ei, ex2, amax2, denom2);
  k_node2_out<<<gbN, TB, 0, stream>>>(ei, rowstart, csr, ex2, denom2, xl2, bias2, out);
}
